// Graphormer3DEncoderLayer_35107062677955
// MI455X (gfx1250) — compile-verified, matrix operand order checked by reading, not run on hardware
//
#include <hip/hip_runtime.h>

typedef __attribute__((ext_vector_type(16))) __bf16 v16bf;
typedef __attribute__((ext_vector_type(8)))  __bf16 v8bf;
typedef __attribute__((ext_vector_type(8)))  float  v8f;

#define E_DIM 768
#define G_DIM 32
#define N_DIM 512
#define H_DIM 8
#define D_DIM 96
#define FF_DIM 3072
#define M_DIM (N_DIM * G_DIM)   // 16384 rows, row r = n*G + g

// CDNA5 async direct-to-LDS copy (ASYNCcnt-tracked), GV addressing mode.
// dst: LDS byte address (low 32 bits of generic pointer), src: 64-bit global addr.
#define ASYNC_LDS_B128(ldsoff, gptr)                                              \
  asm volatile("global_load_async_to_lds_b128 %0, %1, off"                        \
               :: "v"(ldsoff), "v"(gptr) : "memory")
#define WAIT_ASYNC0() asm volatile("s_wait_asynccnt 0x0" ::: "memory")

static __device__ __forceinline__ v16bf pack16(v8bf lo, v8bf hi) {
  v16bf r;
#pragma unroll
  for (int e = 0; e < 8; ++e) { r[e] = lo[e]; r[e + 8] = hi[e]; }
  return r;
}

// ---------------- f32 -> bf16 convert (weights) ----------------
__global__ void cvt_f32_bf16(const float* __restrict__ s, __bf16* __restrict__ d, int n) {
  for (int i = blockIdx.x * blockDim.x + threadIdx.x; i < n; i += gridDim.x * blockDim.x)
    d[i] = (__bf16)s[i];
}

// ---------------- LayerNorm: one wave per row of 768 ----------------
__global__ void __launch_bounds__(256) ln_kernel(const float* __restrict__ x,
                                                 const float* __restrict__ w,
                                                 const float* __restrict__ b,
                                                 __bf16* __restrict__ y) {
  const int lane = threadIdx.x & 31;
  const int row  = blockIdx.x * 8 + (threadIdx.x >> 5);
  const float* xr = x + (size_t)row * E_DIM;
  float v[24];
  float s = 0.f, s2 = 0.f;
#pragma unroll
  for (int j = 0; j < 24; ++j) { float t = xr[lane + 32 * j]; v[j] = t; s += t; s2 += t * t; }
#pragma unroll
  for (int m = 1; m < 32; m <<= 1) { s += __shfl_xor(s, m, 32); s2 += __shfl_xor(s2, m, 32); }
  const float mean = s * (1.f / E_DIM);
  const float var  = s2 * (1.f / E_DIM) - mean * mean;
  const float r    = rsqrtf(var + 1e-5f);
  __bf16* yr = y + (size_t)row * E_DIM;
#pragma unroll
  for (int j = 0; j < 24; ++j) {
    int c = lane + 32 * j;
    yr[c] = (__bf16)((v[j] - mean) * r * w[c] + b[c]);
  }
}

// ---------------- Tiled bf16 GEMM: C[M,Ncol] = A[M,K] @ B[Ncol,K]^T ----------------
// MODE 0: qkv  (scale q cols < 768, out bf16)
// MODE 1: out-proj (+bias, node mask, +residual -> f32)
// MODE 2: fc1 (+bias, exact gelu -> bf16)
// MODE 3: fc2 (+bias, +residual -> f32)
template <int MODE>
__global__ void __launch_bounds__(256) gemm_bf16(const __bf16* __restrict__ A,
                                                 const __bf16* __restrict__ B,
                                                 const float* __restrict__ bias,
                                                 __bf16* __restrict__ outb,
                                                 float* __restrict__ outf,
                                                 const float* __restrict__ resid,
                                                 const unsigned char* __restrict__ mask,
                                                 int Ncol, int K) {
  __shared__ __bf16 As[128 * 40];
  __shared__ __bf16 Bs[128 * 40];
  const int tid  = threadIdx.x;
  const int lane = tid & 31, wave = tid >> 5;
  const int wr = wave & 3, wc = wave >> 2;          // 4x2 wave grid -> 128x128 tile
  const int m0 = blockIdx.y * 128, n0 = blockIdx.x * 128;
  const int coll = lane & 15, sel = lane >> 4;

  // per-thread staging assignments (2 x 16B chunks per operand tile)
  const int c0 = tid, c1 = tid + 256;
  const int row0 = c0 >> 2, ch0 = c0 & 3;
  const int row1 = c1 >> 2, ch1 = c1 & 3;
  const unsigned la0 = (unsigned)(uintptr_t)&As[row0 * 40 + ch0 * 8];
  const unsigned la1 = (unsigned)(uintptr_t)&As[row1 * 40 + ch1 * 8];
  const unsigned lb0 = (unsigned)(uintptr_t)&Bs[row0 * 40 + ch0 * 8];
  const unsigned lb1 = (unsigned)(uintptr_t)&Bs[row1 * 40 + ch1 * 8];

  const v8f z = {0.f, 0.f, 0.f, 0.f, 0.f, 0.f, 0.f, 0.f};
  v8f acc[2][4];
#pragma unroll
  for (int i = 0; i < 2; ++i)
#pragma unroll
    for (int j = 0; j < 4; ++j) acc[i][j] = z;

  for (int kt = 0; kt < K; kt += 32) {
    __syncthreads();  // previous tile fully consumed before overwrite
    ASYNC_LDS_B128(la0, A + (size_t)(m0 + row0) * K + kt + ch0 * 8);
    ASYNC_LDS_B128(la1, A + (size_t)(m0 + row1) * K + kt + ch1 * 8);
    ASYNC_LDS_B128(lb0, B + (size_t)(n0 + row0) * K + kt + ch0 * 8);
    ASYNC_LDS_B128(lb1, B + (size_t)(n0 + row1) * K + kt + ch1 * 8);
    WAIT_ASYNC0();    // this wave's async copies landed in LDS
    __syncthreads();  // all waves' copies landed

    v16bf af[2], bfr[4];
#pragma unroll
    for (int i = 0; i < 2; ++i) {
      int mloc = wr * 32 + 16 * i + coll;
      af[i] = pack16(*(const v8bf*)&As[mloc * 40 + sel * 8],
                     *(const v8bf*)&As[mloc * 40 + 16 + sel * 8]);
    }
#pragma unroll
    for (int j = 0; j < 4; ++j) {
      int nloc = wc * 64 + 16 * j + coll;
      bfr[j] = pack16(*(const v8bf*)&Bs[nloc * 40 + sel * 8],
                      *(const v8bf*)&Bs[nloc * 40 + 16 + sel * 8]);
    }
#pragma unroll
    for (int i = 0; i < 2; ++i)
#pragma unroll
      for (int j = 0; j < 4; ++j)
        acc[i][j] = __builtin_amdgcn_wmma_f32_16x16x32_bf16(
            false, af[i], false, bfr[j], (short)0, acc[i][j], false, false);
  }

  // Epilogue: D layout row = v + 8*(lane>=16), col = lane%16
#pragma unroll
  for (int i = 0; i < 2; ++i) {
#pragma unroll
    for (int j = 0; j < 4; ++j) {
#pragma unroll
      for (int v = 0; v < 8; ++v) {
        int m = m0 + wr * 32 + 16 * i + v + 8 * sel;
        int n = n0 + wc * 64 + 16 * j + coll;
        float val = acc[i][j][v] + bias[n];
        size_t idx = (size_t)m * Ncol + n;
        if (MODE == 0) {
          if (n < E_DIM) val *= 0.1020620726f;  // 96^-0.5 applied to Q
          outb[idx] = (__bf16)val;
        } else if (MODE == 1) {
          int g = m & (G_DIM - 1), node = m >> 5;
          if (!mask[g * N_DIM + node]) val = 0.f;
          outf[idx] = resid[idx] + val;
        } else if (MODE == 2) {
          outb[idx] = (__bf16)(0.5f * val * (1.0f + erff(val * 0.70710678f)));
        } else {
          outf[idx] = resid[idx] + val;
        }
      }
    }
  }
}

// ---------------- Fused flash attention: one wave per 16 query rows ----------------
__global__ void __launch_bounds__(256) attn_kernel(const __bf16* __restrict__ qkv,
                                                   const float* __restrict__ bias,
                                                   __bf16* __restrict__ attn_out) {
  __shared__ __bf16 ldsVt[96 * 40];      // V^T tile: [d][key 0..31], stride 40 keeps 16B align
  __shared__ __bf16 ldsP[8][16 * 40];    // per-wave P re-layout buffer
  const int tid = threadIdx.x, lane = tid & 31, wave = tid >> 5;
  const int b = blockIdx.x, g = b >> 3, h = b & 7;   // b = g*H + h
  const int coll = lane & 15, sel = lane >> 4;
  const int n0 = blockIdx.y * 128 + wave * 16;       // query rows for this wave

  // Q fragments (16 x 96 = 3 x 16x32), Q already scaled in the QKV epilogue
  const __bf16* qrow = qkv + (size_t)((n0 + coll) * G_DIM + g) * (3 * E_DIM) + h * D_DIM;
  v16bf qf[3];
#pragma unroll
  for (int t = 0; t < 3; ++t)
    qf[t] = pack16(*(const v8bf*)(qrow + t * 32 + sel * 8),
                   *(const v8bf*)(qrow + t * 32 + 16 + sel * 8));

  const v8f z = {0.f, 0.f, 0.f, 0.f, 0.f, 0.f, 0.f, 0.f};
  v8f o[6];
#pragma unroll
  for (int j = 0; j < 6; ++j) o[j] = z;
  float mrun[8], lrun[8];
#pragma unroll
  for (int v = 0; v < 8; ++v) { mrun[v] = -1e30f; lrun[v] = 0.f; }

  const size_t brow = (size_t)b * N_DIM * N_DIM;

  for (int kb = 0; kb < 16; ++kb) {      // 16 blocks of 32 keys
    // prefetch next block's bias rows (dominant HBM stream) while this block computes
    if (kb < 15) {
#pragma unroll
      for (int v = 0; v < 8; ++v) {
        int mq = n0 + v + 8 * sel;
        __builtin_prefetch(bias + brow + (size_t)mq * N_DIM + (kb + 1) * 32, 0, 1);
      }
    }
    __syncthreads();
    for (int idx = tid; idx < 32 * 12; idx += 256) {   // stage V^T (32 keys x 96 dims)
      int kr = idx / 12, c = idx % 12;
      v8bf vv = *(const v8bf*)(qkv + (size_t)((kb * 32 + kr) * G_DIM + g) * (3 * E_DIM) +
                               2 * E_DIM + h * D_DIM + c * 8);
#pragma unroll
      for (int e = 0; e < 8; ++e) ldsVt[(c * 8 + e) * 40 + kr] = vv[e];
    }
    __syncthreads();

    // S = Q K^T for 16 rows x 32 keys
    v8f s[2] = {z, z};
#pragma unroll
    for (int j = 0; j < 2; ++j) {
      const __bf16* krow = qkv + (size_t)((kb * 32 + 16 * j + coll) * G_DIM + g) * (3 * E_DIM) +
                           E_DIM + h * D_DIM;
#pragma unroll
      for (int t = 0; t < 3; ++t) {
        v16bf kf = pack16(*(const v8bf*)(krow + t * 32 + sel * 8),
                          *(const v8bf*)(krow + t * 32 + 16 + sel * 8));
        s[j] = __builtin_amdgcn_wmma_f32_16x16x32_bf16(
            false, qf[t], false, kf, (short)0, s[j], false, false);
      }
    }

    // bias add + online softmax (rows live in vgpr v + 8*sel; cols across 16 lanes of a half)
    float xs[2][8];
#pragma unroll
    for (int j = 0; j < 2; ++j)
#pragma unroll
      for (int v = 0; v < 8; ++v) {
        int mq = n0 + v + 8 * sel;
        int nk = kb * 32 + 16 * j + coll;
        xs[j][v] = s[j][v] + bias[brow + (size_t)mq * N_DIM + nk];
      }
    float newm[8], alpha[8];
#pragma unroll
    for (int v = 0; v < 8; ++v) {
      float t = fmaxf(xs[0][v], xs[1][v]);
#pragma unroll
      for (int msk = 1; msk < 16; msk <<= 1) t = fmaxf(t, __shfl_xor(t, msk, 32));
      newm[v]  = fmaxf(mrun[v], t);
      alpha[v] = __expf(mrun[v] - newm[v]);
      mrun[v]  = newm[v];
    }
#pragma unroll
    for (int v = 0; v < 8; ++v) {
      xs[0][v] = __expf(xs[0][v] - newm[v]);
      xs[1][v] = __expf(xs[1][v] - newm[v]);
      float rs = xs[0][v] + xs[1][v];
#pragma unroll
      for (int msk = 1; msk < 16; msk <<= 1) rs += __shfl_xor(rs, msk, 32);
      lrun[v] = lrun[v] * alpha[v] + rs;
    }
#pragma unroll
    for (int jo = 0; jo < 6; ++jo)
#pragma unroll
      for (int v = 0; v < 8; ++v) o[jo][v] = o[jo][v] * alpha[v];

    // D-frag -> A-frag re-layout of P through per-wave LDS (same-wave DS ops are in-order)
    __bf16* pw = &ldsP[wave][0];
#pragma unroll
    for (int j = 0; j < 2; ++j)
#pragma unroll
      for (int v = 0; v < 8; ++v)
        pw[(v + 8 * sel) * 40 + 16 * j + coll] = (__bf16)xs[j][v];

    v16bf pf = pack16(*(const v8bf*)&pw[coll * 40 + sel * 8],
                      *(const v8bf*)&pw[coll * 40 + 16 + sel * 8]);
#pragma unroll
    for (int jo = 0; jo < 6; ++jo) {
      int d = 16 * jo + coll;
      v16bf vf = pack16(*(const v8bf*)&ldsVt[d * 40 + sel * 8],
                        *(const v8bf*)&ldsVt[d * 40 + 16 + sel * 8]);
      o[jo] = __builtin_amdgcn_wmma_f32_16x16x32_bf16(
          false, pf, false, vf, (short)0, o[jo], false, false);
    }
  }

  // normalize + store [N,G,E] at (n, g, h*96 + d)
#pragma unroll
  for (int jo = 0; jo < 6; ++jo)
#pragma unroll
    for (int v = 0; v < 8; ++v) {
      int mq = n0 + v + 8 * sel;
      float val = o[jo][v] / lrun[v];
      attn_out[(size_t)(mq * G_DIM + g) * E_DIM + h * D_DIM + 16 * jo + coll] = (__bf16)val;
    }
}

extern "C" void kernel_launch(void* const* d_in, const int* in_sizes, int n_in,
                              void* d_out, int out_size, void* d_ws, size_t ws_size,
                              hipStream_t stream) {
  const float* query      = (const float*)d_in[0];
  const float* attn_bias  = (const float*)d_in[1];
  const unsigned char* node_mask = (const unsigned char*)d_in[2];
  const float* in_proj_w  = (const float*)d_in[3];
  const float* in_proj_b  = (const float*)d_in[4];
  const float* out_proj_w = (const float*)d_in[5];
  const float* out_proj_b = (const float*)d_in[6];
  const float* ln1_w = (const float*)d_in[7];
  const float* ln1_b = (const float*)d_in[8];
  const float* ln2_w = (const float*)d_in[9];
  const float* ln2_b = (const float*)d_in[10];
  const float* fc1_w = (const float*)d_in[11];
  const float* fc1_b = (const float*)d_in[12];
  const float* fc2_w = (const float*)d_in[13];
  const float* fc2_b = (const float*)d_in[14];
  float* out = (float*)d_out;

  char* ws = (char*)d_ws;
  size_t off = 0;
  auto take = [&](size_t bytes) -> void* {
    void* p = ws + off;
    off += (bytes + 255) & ~(size_t)255;
    return p;
  };
  __bf16* w_in  = (__bf16*)take((size_t)3 * E_DIM * E_DIM * 2);
  __bf16* w_out = (__bf16*)take((size_t)E_DIM * E_DIM * 2);
  __bf16* w_fc1 = (__bf16*)take((size_t)FF_DIM * E_DIM * 2);
  __bf16* w_fc2 = (__bf16*)take((size_t)E_DIM * FF_DIM * 2);
  __bf16* xln   = (__bf16*)take((size_t)M_DIM * E_DIM * 2);
  __bf16* qkv   = (__bf16*)take((size_t)M_DIM * 3 * E_DIM * 2);
  __bf16* attn  = (__bf16*)take((size_t)M_DIM * E_DIM * 2);
  float*  x1    = (float*)take((size_t)M_DIM * E_DIM * 4);
  __bf16* hbuf  = qkv;  // qkv (75.5MB) + attn (25.2MB) regions exactly fit M*FF bf16

  cvt_f32_bf16<<<2048, 256, 0, stream>>>(in_proj_w,  w_in,  3 * E_DIM * E_DIM);
  cvt_f32_bf16<<<2048, 256, 0, stream>>>(out_proj_w, w_out, E_DIM * E_DIM);
  cvt_f32_bf16<<<2048, 256, 0, stream>>>(fc1_w, w_fc1, FF_DIM * E_DIM);
  cvt_f32_bf16<<<2048, 256, 0, stream>>>(fc2_w, w_fc2, E_DIM * FF_DIM);

  ln_kernel<<<M_DIM / 8, 256, 0, stream>>>(query, ln1_w, ln1_b, xln);
  gemm_bf16<0><<<dim3(3 * E_DIM / 128, M_DIM / 128), 256, 0, stream>>>(
      xln, w_in, in_proj_b, qkv, nullptr, nullptr, nullptr, 3 * E_DIM, E_DIM);
  attn_kernel<<<dim3(G_DIM * H_DIM, N_DIM / 128), 256, 0, stream>>>(qkv, attn_bias, attn);
  gemm_bf16<1><<<dim3(E_DIM / 128, M_DIM / 128), 256, 0, stream>>>(
      attn, w_out, out_proj_b, nullptr, x1, query, node_mask, E_DIM, E_DIM);
  ln_kernel<<<M_DIM / 8, 256, 0, stream>>>(x1, ln2_w, ln2_b, xln);
  gemm_bf16<2><<<dim3(FF_DIM / 128, M_DIM / 128), 256, 0, stream>>>(
      xln, w_fc1, fc1_b, hbuf, nullptr, nullptr, nullptr, FF_DIM, E_DIM);
  gemm_bf16<3><<<dim3(E_DIM / 128, M_DIM / 128), 256, 0, stream>>>(
      hbuf, w_fc2, fc2_b, nullptr, out, x1, nullptr, E_DIM, FF_DIM);
}
